// MultiHeadedAttention_34797825032969
// MI455X (gfx1250) — matrix operand order repaired and checked by reading, not compiled, not run on hardware
//
#include <hip/hip_runtime.h>

// ---------------------------------------------------------------------------
// MI455X (gfx1250) fused MHA pipeline.
//  - bf16 WMMA 16x16x32, f32 accumulate (fp32 WMMA is 16x16x4 = 8x less K/inst)
//  - GEMM operand tiles staged by the Tensor Data Mover (tensor_load_to_lds)
//    with hardware LDS row padding, double buffered on TENSORcnt.
//  - Flash attention: scores never leave the WGP (537 MB of S x S avoided);
//    Q/K frags direct from global, V pre-transposed -> zero attn barriers.
// ---------------------------------------------------------------------------

typedef __attribute__((ext_vector_type(16))) __bf16 v16bf;
typedef __attribute__((ext_vector_type(8)))  __bf16 bf16x8;
typedef __attribute__((ext_vector_type(4)))  __bf16 bf16x4;
typedef __attribute__((ext_vector_type(8)))  float  v8f;
typedef __attribute__((ext_vector_type(4)))  float  f32x4;
typedef __attribute__((ext_vector_type(4)))  unsigned int u32x4;
typedef __attribute__((ext_vector_type(8)))  int i32x8;
typedef __attribute__((ext_vector_type(4)))  int i32x4;

#define SHUF16(lo, hi) \
  __builtin_shufflevector((lo), (hi), 0,1,2,3,4,5,6,7,8,9,10,11,12,13,14,15)

static constexpr int Bc  = 2;
static constexpr int Sc  = 2048;
static constexpr int Dc  = 1024;
static constexpr int Hc  = 16;
static constexpr int DKc = 64;
static constexpr int CLc = 256;
static constexpr int HIDc = 512;

// ---------------------------------------------------------------------------
// TDM: 2D strided tile load, global -> LDS, bf16 elements, with LDS padding.
// D# per CDNA5 ISA ch.8: group0 = {count/lds/global/type}, group1 = dims.
// pad_interval=3 (pad every 16 DWORDs = one 32-elem bf16 row),
// pad_amount=3 (4 DWORDs = 16B) -> LDS row stride 80B = 40 bf16.
// ---------------------------------------------------------------------------
__device__ __forceinline__ void tdm_load_2d_bf16(
    unsigned lds_byte_off, const void* gptr,
    unsigned tile_d0, unsigned tile_d1,
    unsigned tensor_d0, unsigned tensor_d1, unsigned stride_d0) {
  const unsigned long long ga = (unsigned long long)(__SIZE_TYPE__)gptr;
  u32x4 g0;
  g0[0] = 1u;                                   // count=1 (valid user D#)
  g0[1] = lds_byte_off;                         // lds_addr
  g0[2] = (unsigned)(ga & 0xffffffffu);         // global_addr[31:0]
  g0[3] = (unsigned)((ga >> 32) & 0x01ffffffu)  // global_addr[56:32]
          | (2u << 30);                         // type=2 ("image")
  i32x8 g1;
  g1[0] = (int)((1u << 16)      // data_size=1 -> 2 bytes
              | (1u << 20)      // pad_enable
              | (3u << 22)      // pad_interval: 16 DWORDs
              | (3u << 25));    // pad_amount: 4 DWORDs
  g1[1] = (int)((tensor_d0 & 0xffffu) << 16);                    // dim0 lo
  g1[2] = (int)((tensor_d0 >> 16) | ((tensor_d1 & 0xffffu) << 16));
  g1[3] = (int)((tensor_d1 >> 16) | (tile_d0 << 16));            // tile_dim0
  g1[4] = (int)(tile_d1 & 0xffffu);                              // tile_dim1
  g1[5] = (int)stride_d0;                                        // stride lo32
  g1[6] = 0;                                                     // stride hi
  g1[7] = 0;
  i32x4 z = {0, 0, 0, 0};
#if defined(__clang_major__) && __clang_major__ >= 23
  i32x8 z8 = {0, 0, 0, 0, 0, 0, 0, 0};
  __builtin_amdgcn_tensor_load_to_lds(g0, g1, z, z, z8, 0);
#else
  __builtin_amdgcn_tensor_load_to_lds(g0, g1, z, z, 0);
#endif
}

// ---------------------------------------------------------------------------
// one-time preprocessing: f32 -> bf16 elementwise (q/k/v activations)
// ---------------------------------------------------------------------------
__global__ __launch_bounds__(256)
void cvt_f32_bf16(const float* __restrict__ x, __bf16* __restrict__ y) {
  const int i = (blockIdx.x * 256 + threadIdx.x) * 4;
  f32x4 v = *(const f32x4*)(x + i);
  bf16x4 o;
  o[0] = (__bf16)v[0]; o[1] = (__bf16)v[1];
  o[2] = (__bf16)v[2]; o[3] = (__bf16)v[3];
  *(bf16x4*)(y + i) = o;
}

// one-time: W[K,N] f32 -> Wt[N,K] bf16 (32x32 LDS tile transpose)
__global__ __launch_bounds__(256)
void transpose_to_bf16(const float* __restrict__ W, __bf16* __restrict__ Wt,
                       int K, int N) {
  __shared__ __bf16 t[32][33];
  const int kb = blockIdx.y * 32, nb = blockIdx.x * 32;
  const int tx = threadIdx.x & 31, ty = threadIdx.x >> 5;   // 32 x 8
  for (int p = 0; p < 4; ++p)
    t[ty + p * 8][tx] = (__bf16)W[(size_t)(kb + ty + p * 8) * N + nb + tx];
  __syncthreads();
  for (int p = 0; p < 4; ++p)
    Wt[(size_t)(nb + ty + p * 8) * K + kb + tx] = t[tx][ty + p * 8];
}

// ---------------------------------------------------------------------------
// cond MLP: c[b, :] = relu(cond[b] @ Wc1 + bc1) @ Wc2   (~1 MFLOP, noise)
// ---------------------------------------------------------------------------
__global__ __launch_bounds__(256)
void cond_mlp_kernel(const float* __restrict__ cond, const float* __restrict__ Wc1,
                     const float* __restrict__ bc1, const float* __restrict__ Wc2,
                     float* __restrict__ cb) {
  __shared__ float hs[HIDc];
  __shared__ float cs[CLc];
  const int b = blockIdx.x, tid = threadIdx.x;
  if (tid < CLc) cs[tid] = cond[b * CLc + tid];
  __syncthreads();
  for (int j = tid; j < HIDc; j += 256) {
    float a = bc1[j];
    for (int i = 0; i < CLc; ++i) a += cs[i] * Wc1[i * HIDc + j];
    hs[j] = fmaxf(a, 0.0f);
  }
  __syncthreads();
  for (int d = tid; d < Dc; d += 256) {
    float a = 0.0f;
    for (int j = 0; j < HIDc; ++j) a += hs[j] * Wc2[j * Dc + d];
    cb[b * Dc + d] = a;
  }
}

// ---------------------------------------------------------------------------
// GEMM: Out[M,N] = A[M,K](bf16) @ Wt[N,K]^T(bf16) + bias[N] (+cbias[b,N])
// Block 128x64, 8 waves (4x2), 2x2 WMMA tiles/wave, BK=32.
// Tiles staged by TDM, double buffered: wave0 issues, s_wait_tensorcnt
// pipelines load(i+1) against compute(i).
// OUT_VT: scatter-store output transposed as [B][H][dk][S] (for V).
// ---------------------------------------------------------------------------
template <bool OUT_BF16, bool ADD_C, bool OUT_VT>
__global__ __launch_bounds__(256)
void gemm_tdm_wmma(const __bf16* __restrict__ A, const __bf16* __restrict__ Wt,
                   const float* __restrict__ bias, const float* __restrict__ cbias,
                   void* __restrict__ Outv, int M, int N, int K) {
  constexpr int BM = 128, BN = 64, BK = 32, AST = 40, BST = 40;
  __shared__ __bf16 sA[2 * BM * AST];
  __shared__ __bf16 sB[2 * BN * BST];
  const int tid  = threadIdx.x;
  const int wave = tid >> 5, lane = tid & 31;
  const int wm = wave >> 1, wn = wave & 1;
  const int l16 = lane & 15, kh = lane >> 4;
  const int bm = blockIdx.y * BM, bn = blockIdx.x * BN;
  const unsigned sA_off = (unsigned)(__SIZE_TYPE__)(void*)&sA[0];
  const unsigned sB_off = (unsigned)(__SIZE_TYPE__)(void*)&sB[0];

  v8f acc[2][2];
  for (int i = 0; i < 2; ++i)
    for (int j = 0; j < 2; ++j)
      for (int e = 0; e < 8; ++e) acc[i][j][e] = 0.0f;

  // prologue: DMA first tiles into buffer 0
  if (wave == 0) {
    tdm_load_2d_bf16(sA_off, A + (size_t)bm * K, BK, BM, K, M, K);
    tdm_load_2d_bf16(sB_off, Wt + (size_t)bn * K, BK, BN, K, N, K);
  }

  int buf = 0;
  for (int k0 = 0; k0 < K; k0 += BK) {
    __syncthreads();   // everyone done reading buf^1 from previous iteration
    if (wave == 0) {
      if (k0 + BK < K) {   // prefetch next tiles into the other buffer
        const unsigned boff = (unsigned)(buf ^ 1);
        tdm_load_2d_bf16(sA_off + boff * BM * AST * 2,
                         A + (size_t)bm * K + (k0 + BK), BK, BM, K, M, K);
        tdm_load_2d_bf16(sB_off + boff * BN * BST * 2,
                         Wt + (size_t)bn * K + (k0 + BK), BK, BN, K, N, K);
        __builtin_amdgcn_s_wait_tensorcnt(2);  // current 2 loads complete
      } else {
        __builtin_amdgcn_s_wait_tensorcnt(0);
      }
    }
    __syncthreads();   // release compute on current buffer

    const __bf16* cA = &sA[buf * BM * AST];
    const __bf16* cB = &sB[buf * BN * BST];
    for (int mi = 0; mi < 2; ++mi) {
      const int row = wm * 32 + mi * 16 + l16;
      // A-frag (16x32): lane<16 -> K {0..7,16..23}; lane>=16 -> {8..15,24..31}
      bf16x8 alo = *(const bf16x8*)(&cA[row * AST + kh * 8]);
      bf16x8 ahi = *(const bf16x8*)(&cA[row * AST + 16 + kh * 8]);
      v16bf af = SHUF16(alo, ahi);
      for (int ni = 0; ni < 2; ++ni) {
        const int col = wn * 32 + ni * 16 + l16;
        // B-frag (32x16): lane<16 -> K {0..7,16..23}; lane>=16 -> {8..15,24..31}
        bf16x8 blo = *(const bf16x8*)(&cB[col * BST + kh * 8]);
        bf16x8 bhi = *(const bf16x8*)(&cB[col * BST + 16 + kh * 8]);
        v16bf bfr = SHUF16(blo, bhi);
        acc[mi][ni] = __builtin_amdgcn_wmma_f32_16x16x32_bf16(
            false, af, false, bfr, (short)0, acc[mi][ni], false, false);
      }
    }
    buf ^= 1;
  }

  // epilogue: C-layout (lane<16: M=r, lane>=16: M=r+8; N = lane%16)
  for (int mi = 0; mi < 2; ++mi)
    for (int ni = 0; ni < 2; ++ni) {
      const int col = bn + wn * 32 + ni * 16 + l16;
      const int rb  = bm + wm * 32 + mi * 16 + kh * 8;
      const float bcol = bias[col];
      for (int r = 0; r < 8; ++r) {
        const int row = rb + r;
        float v = acc[mi][ni][r] + bcol;
        if (ADD_C) v += cbias[(row >> 11) * Dc + col];   // row/2048 == batch
        if (OUT_VT) {
          const int h = col >> 6, dk = col & 63;
          ((__bf16*)Outv)[(((size_t)(row >> 11) * Hc + h) * DKc + dk) * Sc +
                          (row & (Sc - 1))] = (__bf16)v;
        } else if (OUT_BF16) {
          ((__bf16*)Outv)[(size_t)row * N + col] = (__bf16)v;
        } else {
          ((float*)Outv)[(size_t)row * N + col] = v;
        }
      }
    }
}

// ---------------------------------------------------------------------------
// Flash attention: grid (S/128, B*H). 8 waves/block, 16 q-rows per wave.
// Q/K frags direct from global (contiguous per lane); V pre-transposed
// [B][H][dk][S] -> P@V B-frags are contiguous too. No __syncthreads at all;
// only per-wave LDS round trip for P (C-layout -> A-layout).
// ---------------------------------------------------------------------------
__global__ __launch_bounds__(256)
void attn_flash_wmma(const __bf16* __restrict__ Q, const __bf16* __restrict__ K,
                     const __bf16* __restrict__ Vt, __bf16* __restrict__ X) {
  constexpr int KT = 32, PST = 40;
  __shared__ __bf16 sP[8 * 16 * PST];     // per-wave P scratch (16 x 32)
  const int tid = threadIdx.x, wave = tid >> 5, lane = tid & 31;
  const int l16 = lane & 15, kh = lane >> 4;
  const int bh = blockIdx.y, b = bh >> 4, h = bh & 15;
  const int qbase = blockIdx.x * 128 + wave * 16;   // row within batch
  const __bf16* Qb = Q + (size_t)b * Sc * Dc + h * DKc;
  const __bf16* Kb = K + (size_t)b * Sc * Dc + h * DKc;
  const __bf16* Vb = Vt + ((size_t)(b * Hc + h)) * DKc * Sc;  // [dk][s]

  // Q A-frags for dk 0..31 and 32..63 (persist whole kernel)
  v16bf qf[2];
  for (int ks = 0; ks < 2; ++ks) {
    const __bf16* p = Qb + (size_t)(qbase + l16) * Dc + ks * 32 + kh * 8;
    bf16x8 lo = *(const bf16x8*)p;
    bf16x8 hi = *(const bf16x8*)(p + 16);
    qf[ks] = SHUF16(lo, hi);
  }

  v8f ao[4];
  for (int nt = 0; nt < 4; ++nt)
    for (int e = 0; e < 8; ++e) ao[nt][e] = 0.0f;
  float m_i[8], l_i[8];
  for (int r = 0; r < 8; ++r) { m_i[r] = -1e30f; l_i[r] = 0.0f; }
  __bf16* myP = &sP[wave * 16 * PST];
  const float scale = 0.125f;             // 1/sqrt(64)

  for (int kt = 0; kt < Sc; kt += KT) {
    // scores S = Q @ K^T : two 16x16 key tiles, contraction dk=64 (2 ksteps)
    v8f sc0, sc1;
    for (int e = 0; e < 8; ++e) { sc0[e] = 0.0f; sc1[e] = 0.0f; }
    for (int ks = 0; ks < 2; ++ks) {
      const __bf16* p0 = Kb + (size_t)(kt + l16) * Dc + ks * 32 + kh * 8;
      bf16x8 l0 = *(const bf16x8*)p0;
      bf16x8 h0 = *(const bf16x8*)(p0 + 16);
      v16bf kf0 = SHUF16(l0, h0);
      sc0 = __builtin_amdgcn_wmma_f32_16x16x32_bf16(
          false, qf[ks], false, kf0, (short)0, sc0, false, false);
      const __bf16* p1 = Kb + (size_t)(kt + 16 + l16) * Dc + ks * 32 + kh * 8;
      bf16x8 l1 = *(const bf16x8*)p1;
      bf16x8 h1 = *(const bf16x8*)(p1 + 16);
      v16bf kf1 = SHUF16(l1, h1);
      sc1 = __builtin_amdgcn_wmma_f32_16x16x32_bf16(
          false, qf[ks], false, kf1, (short)0, sc1, false, false);
    }

    // online softmax: rows live in 16-lane halves (C-layout)
    float pa[8], pb[8], corr[8];
    for (int r = 0; r < 8; ++r) {
      const float s0 = sc0[r] * scale, s1 = sc1[r] * scale;
      float t = fmaxf(s0, s1);
      t = fmaxf(t, __shfl_xor(t, 1, 32));
      t = fmaxf(t, __shfl_xor(t, 2, 32));
      t = fmaxf(t, __shfl_xor(t, 4, 32));
      t = fmaxf(t, __shfl_xor(t, 8, 32));
      const float mn = fmaxf(m_i[r], t);
      corr[r] = __expf(m_i[r] - mn);
      pa[r] = __expf(s0 - mn);
      pb[r] = __expf(s1 - mn);
      float rs = pa[r] + pb[r];
      rs += __shfl_xor(rs, 1, 32);
      rs += __shfl_xor(rs, 2, 32);
      rs += __shfl_xor(rs, 4, 32);
      rs += __shfl_xor(rs, 8, 32);
      l_i[r] = l_i[r] * corr[r] + rs;
      m_i[r] = mn;
    }
    for (int nt = 0; nt < 4; ++nt)
      for (int r = 0; r < 8; ++r) ao[nt][r] *= corr[r];

    // C-layout P -> per-wave LDS -> A-layout frag (16 q x 32 keys)
    for (int r = 0; r < 8; ++r) {
      const int prow = r + kh * 8;
      myP[prow * PST + l16]      = (__bf16)pa[r];
      myP[prow * PST + 16 + l16] = (__bf16)pb[r];
    }
    asm volatile("s_wait_dscnt 0" ::: "memory");   // wave-local LDS RAW
    bf16x8 plo = *(const bf16x8*)(&myP[l16 * PST + kh * 8]);
    bf16x8 phi = *(const bf16x8*)(&myP[l16 * PST + 16 + kh * 8]);
    v16bf pf = SHUF16(plo, phi);

    // acc += P (16x32) @ V (32x64): B-frag direct from transposed V
    for (int nt = 0; nt < 4; ++nt) {
      const __bf16* vp = Vb + (size_t)(nt * 16 + l16) * Sc + kt + kh * 8;
      bf16x8 vlo = *(const bf16x8*)vp;
      bf16x8 vhi = *(const bf16x8*)(vp + 16);
      v16bf vf = SHUF16(vlo, vhi);
      ao[nt] = __builtin_amdgcn_wmma_f32_16x16x32_bf16(
          false, pf, false, vf, (short)0, ao[nt], false, false);
    }
  }

  // normalize and store x (bf16, row-major [B,S,D]) for the output GEMM
  for (int nt = 0; nt < 4; ++nt) {
    const int col = h * DKc + nt * 16 + l16;
    for (int r = 0; r < 8; ++r) {
      const int row = qbase + r + kh * 8;
      const float o = ao[nt][r] / l_i[r];
      X[((size_t)b * Sc + row) * Dc + col] = (__bf16)o;
    }
  }
}

// ---------------------------------------------------------------------------
extern "C" void kernel_launch(void* const* d_in, const int* in_sizes, int n_in,
                              void* d_out, int out_size, void* d_ws, size_t ws_size,
                              hipStream_t stream) {
  (void)in_sizes; (void)n_in; (void)out_size; (void)ws_size;
  const float* query = (const float*)d_in[0];
  const float* key_  = (const float*)d_in[1];
  const float* value = (const float*)d_in[2];
  const float* cond  = (const float*)d_in[3];
  const float* Wq  = (const float*)d_in[4];  const float* bq  = (const float*)d_in[5];
  const float* Wk  = (const float*)d_in[6];  const float* bk  = (const float*)d_in[7];
  const float* Wv  = (const float*)d_in[8];  const float* bv  = (const float*)d_in[9];
  const float* Wo  = (const float*)d_in[10]; const float* bo  = (const float*)d_in[11];
  const float* Wc1 = (const float*)d_in[12]; const float* bc1 = (const float*)d_in[13];
  const float* Wc2 = (const float*)d_in[14];
  float* out = (float*)d_out;

  // workspace layout (256B aligned slices)
  char* ws = (char*)d_ws;
  size_t off = 0;
  auto take = [&](size_t bytes) -> void* {
    void* p = ws + off;
    off = (off + bytes + 255) & ~(size_t)255;
    return p;
  };
  const size_t ACT = (size_t)Bc * Sc * Dc;      // 4M elements
  float*  cb  = (float*)take((size_t)Bc * Dc * sizeof(float));
  __bf16* qbf = (__bf16*)take(ACT * 2);
  __bf16* kbf = (__bf16*)take(ACT * 2);
  __bf16* vbf = (__bf16*)take(ACT * 2);
  __bf16* WqT = (__bf16*)take((size_t)Dc * Dc * 2);
  __bf16* WkT = (__bf16*)take((size_t)Dc * Dc * 2);
  __bf16* WvT = (__bf16*)take((size_t)Dc * Dc * 2);
  __bf16* WoT = (__bf16*)take((size_t)Dc * Dc * 2);
  __bf16* Qw  = (__bf16*)take(ACT * 2);
  __bf16* Kw  = (__bf16*)take(ACT * 2);
  __bf16* Vtw = (__bf16*)take(ACT * 2);
  __bf16* Xw  = qbf;   // alias: qbf is dead after the Q projection GEMM

  const int M = Bc * Sc, N = Dc, K = Dc;
  dim3 blk(256);
  dim3 gg(N / 64, M / 128);
  dim3 gt(Dc / 32, Dc / 32);

  // preprocessing: bf16 activations + transposed bf16 weights (one-time)
  cvt_f32_bf16<<<dim3((unsigned)(ACT / 1024)), blk, 0, stream>>>(query, qbf);
  cvt_f32_bf16<<<dim3((unsigned)(ACT / 1024)), blk, 0, stream>>>(key_,  kbf);
  cvt_f32_bf16<<<dim3((unsigned)(ACT / 1024)), blk, 0, stream>>>(value, vbf);
  transpose_to_bf16<<<gt, blk, 0, stream>>>(Wq, WqT, Dc, Dc);
  transpose_to_bf16<<<gt, blk, 0, stream>>>(Wk, WkT, Dc, Dc);
  transpose_to_bf16<<<gt, blk, 0, stream>>>(Wv, WvT, Dc, Dc);
  transpose_to_bf16<<<gt, blk, 0, stream>>>(Wo, WoT, Dc, Dc);
  cond_mlp_kernel<<<dim3(Bc), blk, 0, stream>>>(cond, Wc1, bc1, Wc2, cb);

  // projections (TDM-staged WMMA GEMMs)
  gemm_tdm_wmma<true,  false, false><<<gg, blk, 0, stream>>>(qbf, WqT, bq, nullptr, Qw,  M, N, K);
  gemm_tdm_wmma<true,  false, false><<<gg, blk, 0, stream>>>(kbf, WkT, bk, nullptr, Kw,  M, N, K);
  gemm_tdm_wmma<true,  true,  true ><<<gg, blk, 0, stream>>>(vbf, WvT, bv, cb,      Vtw, M, N, K);

  // flash attention
  attn_flash_wmma<<<dim3(Sc / 128, Bc * Hc), blk, 0, stream>>>(Qw, Kw, Vtw, Xw);

  // output projection (f32 result)
  gemm_tdm_wmma<false, false, false><<<gg, blk, 0, stream>>>(Xw, WoT, bo, nullptr, out, M, N, K);
}
